// textLSTM_90915867721771
// MI455X (gfx1250) — matrix operand order repaired and checked by reading, not compiled, not run on hardware
//
#include <hip/hip_runtime.h>

#define V_  50000
#define E_  512
#define H_  1024
#define B_  128
#define T_  512
#define O_  10
#define G4_ 4096
#define NWG_ 64   // persistent workgroups in recurrent kernel (one per WGP, 256KB LDS each)

typedef __attribute__((ext_vector_type(16))) __bf16          v16bf;
typedef __attribute__((ext_vector_type(8)))  float           v8f;
typedef __attribute__((ext_vector_type(8)))  unsigned short  v8us;
typedef __attribute__((ext_vector_type(4)))  unsigned int    v4u;
typedef __attribute__((ext_vector_type(8)))  int             v8i;
typedef __attribute__((ext_vector_type(4)))  int             v4i;

union FragAB { v16bf bf; v8us us2[2]; unsigned short us[16]; };

__device__ inline unsigned short f2bf(float f) {
  unsigned u = __float_as_uint(f);
  u += 0x7fffu + ((u >> 16) & 1u);       // round-to-nearest-even
  return (unsigned short)(u >> 16);
}
__device__ inline float bf2f(unsigned short s) {
  return __uint_as_float(((unsigned)s) << 16);
}
__device__ inline float sigm_(float x) { return 1.0f / (1.0f + __expf(-x)); }
__device__ inline float tanh_(float x) { float e = __expf(-2.0f * x); return (1.0f - e) / (1.0f + e); }

// ---------------------------------------------------------------- init state
__global__ void k_init(unsigned short* hbf, unsigned int* cnt) {
  int i = blockIdx.x * blockDim.x + threadIdx.x;
  if (i < B_ * H_) hbf[i] = 0;
  if (i < 2 * T_)  cnt[i] = 0u;
}

// ------------------------------------------------- pack weights -> bf16 [N][K]
__global__ void k_pack(const float* Wxi, const float* Wxf, const float* Wxo, const float* Wxc,
                       const float* Whi, const float* Whf, const float* Who, const float* Whc,
                       unsigned short* WxT, unsigned short* WhT) {
  long long i = (long long)blockIdx.x * blockDim.x + threadIdx.x;
  if (i < (long long)G4_ * E_) {
    int n = (int)(i / E_), k = (int)(i % E_);
    int gate = n >> 10, col = n & (H_ - 1);
    const float* W = gate == 0 ? Wxi : gate == 1 ? Wxf : gate == 2 ? Wxo : Wxc;
    WxT[(size_t)n * E_ + k] = f2bf(W[(size_t)k * H_ + col]);
  } else {
    long long j = i - (long long)G4_ * E_;
    if (j < (long long)G4_ * H_) {
      int n = (int)(j / H_), k = (int)(j % H_);
      int gate = n >> 10, col = n & (H_ - 1);
      const float* W = gate == 0 ? Whi : gate == 1 ? Whf : gate == 2 ? Who : Whc;
      WhT[(size_t)n * H_ + k] = f2bf(W[(size_t)k * H_ + col]);
    }
  }
}

// ---------------------------------------------- embedding gather -> bf16 rows
__global__ void k_gather(const int* ids, const float* emb, unsigned short* x) {
  int row = blockIdx.x;                       // b*T + t
  int id  = ids[row];
  const float4* src = (const float4*)(emb + (size_t)id * E_);
  unsigned short* dst = x + (size_t)row * E_ + threadIdx.x * 4;
  float4 v = src[threadIdx.x];
  dst[0] = f2bf(v.x); dst[1] = f2bf(v.y); dst[2] = f2bf(v.z); dst[3] = f2bf(v.w);
}

// --------------------------- xproj[t][b][4H] = x @ [Wxi|Wxf|Wxo|Wxc] + bias (f32)
__global__ void __launch_bounds__(256) k_xproj(const unsigned short* x, const unsigned short* WxT,
                                               const float* bi, const float* bfv,
                                               const float* bo, const float* bc, float* xproj) {
  int wave = threadIdx.x >> 5;
  int lane = threadIdx.x & 31;
  int nl = lane & 15, hi = lane >> 4;
  unsigned bid = blockIdx.x;
  int t     = bid >> 8;                 // 256 blocks per t
  int rem   = bid & 255;
  int btile = rem >> 5;                 // 8 row tiles
  int ntile = ((rem & 31) << 3) + wave; // 256 col tiles, 8 per block
  int ncol  = (ntile << 4) + nl;

  float bias = (ncol < H_)     ? bi[ncol]
             : (ncol < 2 * H_) ? bfv[ncol - H_]
             : (ncol < 3 * H_) ? bo[ncol - 2 * H_]
                               : bc[ncol - 3 * H_];
  v8f acc = {bias, bias, bias, bias, bias, bias, bias, bias};

  const unsigned short* arow = x   + ((size_t)(btile * 16 + nl) * T_ + t) * E_ + hi * 8;
  const unsigned short* brow = WxT + (size_t)ncol * E_ + hi * 8;
#pragma unroll 4
  for (int k = 0; k < E_; k += 32) {
    FragAB a, b;
    a.us2[0] = *(const v8us*)(arow + k);
    a.us2[1] = *(const v8us*)(arow + k + 16);
    b.us2[0] = *(const v8us*)(brow + k);
    b.us2[1] = *(const v8us*)(brow + k + 16);
    acc = __builtin_amdgcn_wmma_f32_16x16x32_bf16(false, a.bf, false, b.bf,
                                                  (short)0, acc, false, false);
  }
  float* out = xproj + (size_t)t * (B_ * G4_) + (size_t)(btile * 16 + hi * 8) * G4_ + ncol;
#pragma unroll
  for (int r = 0; r < 8; r++) out[(size_t)r * G4_] = acc[r];
}

// -------------------------------------- persistent cooperative LSTM recurrence
__global__ void __launch_bounds__(256) k_lstm(const unsigned short* WhT, const float* xproj,
                                              unsigned short* hbf, unsigned int* cnt) {
  extern __shared__ unsigned short hsh[];   // full h_t: [128][1024] bf16 = 256 KB
  const int wg   = blockIdx.x;              // owns h columns [wg*16, wg*16+16)
  const int wave = threadIdx.x >> 5;        // btile 0..7
  const int lane = threadIdx.x & 31;
  const int nl = lane & 15, hi = lane >> 4;
  const int m0 = wave * 16;
  const int hc = wg * 16 + nl;              // this lane's h column

  float c[8];
#pragma unroll
  for (int r = 0; r < 8; r++) c[r] = 0.f;   // cell state lives in registers

  // TDM 2D descriptor: load h [128 rows x 1024 cols x 2B] into LDS offset 0
  unsigned long long ga = (unsigned long long)(const void*)hbf;
  v4u g0 = { 1u,                                   // count=1 valid descriptor
             0u,                                   // lds_addr = 0
             (unsigned)(ga & 0xffffffffu),         // global_addr[31:0]
             0x80000000u | (unsigned)((ga >> 32) & 0x01ffffffu) }; // type=2 | addr[56:32]
  v8i g1 = { 0x00010000,   // data_size=1 (2B)
             0x04000000,   // tensor_dim0[15:0]=1024 in bits[63:48]
             0x00800000,   // tensor_dim1[15:0]=128  in bits[95:80]
             0x04000000,   // tile_dim0=1024 in bits[127:112]
             0x00000080,   // tile_dim1=128
             0x00000400,   // tensor_dim0_stride=1024
             0, 0 };
  v4i gz4 = {0, 0, 0, 0};
  v8i gz8 = {0, 0, 0, 0, 0, 0, 0, 0};

  for (int t = 0; t < T_; ++t) {
    if (wave == 0) {
      __builtin_amdgcn_tensor_load_to_lds(g0, g1, gz4, gz4, gz8, 0);
      __builtin_amdgcn_s_wait_tensorcnt(0);
    }
    __syncthreads();
    // grid barrier A: all workgroups captured h_t before anyone overwrites it
    if (threadIdx.x == 0) {
      __hip_atomic_fetch_add(&cnt[2 * t], 1u, __ATOMIC_ACQ_REL, __HIP_MEMORY_SCOPE_AGENT);
      while (__hip_atomic_load(&cnt[2 * t], __ATOMIC_ACQUIRE, __HIP_MEMORY_SCOPE_AGENT) < NWG_)
        __builtin_amdgcn_s_sleep(1);
    }
    __syncthreads();

    // C := xproj[t] tile (bias already folded in)
    const float* xp = xproj + (size_t)t * (B_ * G4_) + (size_t)(m0 + hi * 8) * G4_;
    v8f acc[4];
#pragma unroll
    for (int g = 0; g < 4; ++g) {
      int n0 = (g << 10) + hc;
#pragma unroll
      for (int r = 0; r < 8; r++) acc[g][r] = xp[(size_t)r * G4_ + n0];
    }
    if (t + 1 < T_)
      __builtin_prefetch(xproj + (size_t)(t + 1) * (B_ * G4_) + (size_t)(m0 + hi * 8) * G4_ + hc, 0, 0);

    // acc[g] += h_t(btile) @ Wh[:, gate cols]   (A from LDS, B from L2-resident WhT)
#pragma unroll 1
    for (int k = 0; k < H_; k += 32) {
      FragAB a;
      const unsigned short* ap = hsh + (size_t)(m0 + nl) * H_ + k + hi * 8;
      a.us2[0] = *(const v8us*)(ap);
      a.us2[1] = *(const v8us*)(ap + 16);
#pragma unroll
      for (int g = 0; g < 4; ++g) {
        FragAB b;
        const unsigned short* bp = WhT + (size_t)((g << 10) + (wg << 4) + nl) * H_ + k + hi * 8;
        b.us2[0] = *(const v8us*)(bp);
        b.us2[1] = *(const v8us*)(bp + 16);
        acc[g] = __builtin_amdgcn_wmma_f32_16x16x32_bf16(false, a.bf, false, b.bf,
                                                         (short)0, acc[g], false, false);
      }
    }

    // gates + state update, h_{t+1} chunk to global (bf16)
#pragma unroll
    for (int r = 0; r < 8; r++) {
      float iv = sigm_(acc[0][r]);
      float fv = sigm_(acc[1][r]);
      float ov = sigm_(acc[2][r]);
      float gv = tanh_(acc[3][r]);
      c[r] = fv * c[r] + iv * gv;
      float hv = ov * tanh_(c[r]);
      hbf[(size_t)(m0 + hi * 8 + r) * H_ + hc] = f2bf(hv);
    }
    __threadfence();
    __syncthreads();
    // grid barrier B: all h_{t+1} stores visible before next step's TDM loads
    if (threadIdx.x == 0) {
      __hip_atomic_fetch_add(&cnt[2 * t + 1], 1u, __ATOMIC_ACQ_REL, __HIP_MEMORY_SCOPE_AGENT);
      while (__hip_atomic_load(&cnt[2 * t + 1], __ATOMIC_ACQUIRE, __HIP_MEMORY_SCOPE_AGENT) < NWG_)
        __builtin_amdgcn_s_sleep(1);
    }
    __syncthreads();
  }
}

// ------------------------------------------------ output head: softmax(h@Wq+b)
__global__ void k_head(const unsigned short* hbf, const float* Whq, const float* bq, float* out) {
  int b = threadIdx.x;  // 128 threads
  float acc[O_];
#pragma unroll
  for (int o = 0; o < O_; o++) acc[o] = bq[o];
  const unsigned short* hr = hbf + (size_t)b * H_;
  for (int k = 0; k < H_; k++) {
    float hv = bf2f(hr[k]);
#pragma unroll
    for (int o = 0; o < O_; o++) acc[o] += hv * Whq[(size_t)k * O_ + o];
  }
  float mx = acc[0];
  for (int o = 1; o < O_; o++) mx = fmaxf(mx, acc[o]);
  float s = 0.f;
  for (int o = 0; o < O_; o++) { acc[o] = __expf(acc[o] - mx); s += acc[o]; }
  float inv = 1.0f / s;
  for (int o = 0; o < O_; o++) out[b * O_ + o] = acc[o] * inv;
}

extern "C" void kernel_launch(void* const* d_in, const int* in_sizes, int n_in,
                              void* d_out, int out_size, void* d_ws, size_t ws_size,
                              hipStream_t stream) {
  (void)in_sizes; (void)n_in; (void)out_size; (void)ws_size;
  const int*   ids = (const int*)  d_in[0];
  const float* emb = (const float*)d_in[1];
  const float* Wxi = (const float*)d_in[2];
  const float* Whi = (const float*)d_in[3];
  const float* bi  = (const float*)d_in[4];
  const float* Wxf = (const float*)d_in[5];
  const float* Whf = (const float*)d_in[6];
  const float* bfv = (const float*)d_in[7];
  const float* Wxo = (const float*)d_in[8];
  const float* Who = (const float*)d_in[9];
  const float* bo  = (const float*)d_in[10];
  const float* Wxc = (const float*)d_in[11];
  const float* Whc = (const float*)d_in[12];
  const float* bc  = (const float*)d_in[13];
  const float* Whq = (const float*)d_in[14];
  const float* bq  = (const float*)d_in[15];

  char* ws = (char*)d_ws;
  size_t off = 0;
  auto take = [&](size_t bytes) -> char* {
    char* p = ws + off;
    off += (bytes + 255) & ~(size_t)255;
    return p;
  };
  unsigned short* x     = (unsigned short*)take((size_t)B_ * T_ * E_ * 2);   //  64 MB
  unsigned short* WxT   = (unsigned short*)take((size_t)G4_ * E_ * 2);       //   4 MB
  unsigned short* WhT   = (unsigned short*)take((size_t)G4_ * H_ * 2);       //   8 MB
  unsigned short* hbf   = (unsigned short*)take((size_t)B_ * H_ * 2);        // 256 KB
  unsigned int*   cnt   = (unsigned int*)  take((size_t)2 * T_ * 4);         //   4 KB
  float*          xproj = (float*)         take((size_t)T_ * B_ * G4_ * 4);  //   1 GB

  k_init<<<(B_ * H_ + 255) / 256, 256, 0, stream>>>(hbf, cnt);

  long long packTot = (long long)G4_ * E_ + (long long)G4_ * H_;
  k_pack<<<(unsigned)((packTot + 255) / 256), 256, 0, stream>>>(
      Wxi, Wxf, Wxo, Wxc, Whi, Whf, Who, Whc, WxT, WhT);

  k_gather<<<B_ * T_, 128, 0, stream>>>(ids, emb, x);

  k_xproj<<<T_ * 256, 256, 0, stream>>>(x, WxT, bi, bfv, bo, bc, xproj);

  k_lstm<<<NWG_, 256, (size_t)B_ * H_ * 2, stream>>>(WhT, xproj, hbf, cnt);

  k_head<<<1, 128, 0, stream>>>(hbf, Whq, bq, (float*)d_out);
}
